// TwoStageCaptchaModel_91302414778902
// MI455X (gfx1250) — matrix operand order repaired and checked by reading, not compiled, not run on hardware
//
#include <hip/hip_runtime.h>
#include <hip/hip_bf16.h>
#include <cstdint>

typedef _Float16 h16;
typedef __attribute__((ext_vector_type(16))) _Float16 v16h;
typedef __attribute__((ext_vector_type(8)))  float    v8f;

#define BN_EPS 1e-5f

// ---------------------------------------------------------------------------
// Workspace layout (bytes, all offsets multiples of 256)
// ---------------------------------------------------------------------------
static const size_t OFF_P0  = 0;                     // ping buffer   (39.32 MB)
static const size_t SZ_P0   = 39321600;
static const size_t OFF_P1  = OFF_P0 + SZ_P0;        // pong buffer   (19.66 MB)
static const size_t SZ_P1   = 19660800;
static const size_t OFF_R   = OFF_P1 + SZ_P1;        // conv-out      (78.64 MB)
static const size_t SZ_R    = 78643200;
static const size_t OFF_WT  = OFF_R + SZ_R;          // f16 weight tiles
static const size_t WT_DC2  = OFF_WT;                //    36,864 B
static const size_t WT_DC3  = WT_DC2 + 36864;        //   147,456 B
static const size_t WT_DC4  = WT_DC3 + 147456;       //   589,824 B
static const size_t WT_DFC  = WT_DC4 + 589824;       // 7,864,320 B
static const size_t WT_DHD  = WT_DFC + 7864320;      //    32,768 B
static const size_t WT_RC2  = WT_DHD + 32768;        //    36,864 B
static const size_t WT_RC3  = WT_RC2 + 36864;        //   147,456 B
static const size_t WT_RFC  = WT_RC3 + 147456;       // 1,638,400 B
static const size_t WT_RHD  = WT_RFC + 1638400;      //    24,576 B
static const size_t OFF_SC  = WT_RHD + 24576;        // folded BN scales (16 KB)
static const size_t OFF_SH  = OFF_SC + 16384;        // folded BN shifts (16 KB)
static const size_t OFF_BB  = OFF_SH + 16384;        // raw det head  (25,600 B)
static const size_t OFF_BB2 = OFF_BB + 25600;        // bboxes f32    (25,600 B)

// ---------------------------------------------------------------------------
// Prep: fold BN(+bias) into per-channel scale/shift
// ---------------------------------------------------------------------------
__global__ void fold_bn_kernel(const float* __restrict__ g, const float* __restrict__ b,
                               const float* __restrict__ m, const float* __restrict__ v,
                               const float* __restrict__ bias,
                               float* __restrict__ scale, float* __restrict__ shift, int C)
{
    int i = blockIdx.x * blockDim.x + threadIdx.x;
    if (i >= C) return;
    float s = g[i] * rsqrtf(v[i] + BN_EPS);
    scale[i] = s;
    shift[i] = b[i] - m[i] * s + bias[i] * s;
}

__global__ void head_affine_kernel(const float* __restrict__ bias,
                                   float* __restrict__ scale, float* __restrict__ shift,
                                   int Creal, int Cpad)
{
    int i = blockIdx.x * blockDim.x + threadIdx.x;
    if (i >= Cpad) return;
    scale[i] = 1.0f;
    shift[i] = (i < Creal) ? bias[i] : 0.0f;
}

// ---------------------------------------------------------------------------
// Prep: tile f32 weights into a WMMA A-fragment layout (contiguous k, same order as B), f16.
//   linear element e = ((kc*Mtiles + mt)*32 + lane)*16 + i
//   row  M = mt*16 + lane%16
//   col  K = kc*32 + i + 16*(lane>>4)
// ---------------------------------------------------------------------------
__global__ void tile_conv_w_kernel(const float* __restrict__ w, h16* __restrict__ out,
                                   int Cin, int Cout)
{
    int K = Cin * 9, Mt = Cout >> 4;
    int total = (K >> 5) * Mt * 512;
    int e = blockIdx.x * blockDim.x + threadIdx.x;
    if (e >= total) return;
    int i = e & 15, L = (e >> 4) & 31, t = e >> 9;
    int mt = t % Mt, kc = t / Mt;
    int mrow = mt * 16 + (L & 15);
    int k = kc * 32 + i + 16 * (L >> 4);
    int ci = k / 9, r = k - ci * 9, ky = r / 3, kx = r - ky * 3;
    out[e] = (h16)w[((mrow * Cin + ci) * 3 + ky) * 3 + kx];
}

__global__ void tile_fc_w_kernel(const float* __restrict__ w, h16* __restrict__ out,
                                 int K, int Mreal, int Mp)
{
    int Mt = Mp >> 4;
    int total = (K >> 5) * Mt * 512;
    int e = blockIdx.x * blockDim.x + threadIdx.x;
    if (e >= total) return;
    int i = e & 15, L = (e >> 4) & 31, t = e >> 9;
    int mt = t % Mt, kc = t / Mt;
    int mrow = mt * 16 + (L & 15);
    int k = kc * 32 + i + 16 * (L >> 4);
    out[e] = (mrow < Mreal) ? (h16)w[(int64_t)mrow * K + k] : (h16)0.0f;
}

// ---------------------------------------------------------------------------
// Cin==1 conv (K=9): direct fused conv+BN+ReLU+2x2 maxpool
// ---------------------------------------------------------------------------
template <typename T>
__global__ void conv1_pool_kernel(const T* __restrict__ in, const float* __restrict__ w9,
                                  const float* __restrict__ scale, const float* __restrict__ shift,
                                  h16* __restrict__ out,
                                  int N, int Cout, int H, int W, int Ho, int Wo)
{
    int64_t idx = (int64_t)blockIdx.x * blockDim.x + threadIdx.x;
    int64_t total = (int64_t)N * Cout * Ho * Wo;
    if (idx >= total) return;
    int px = (int)(idx % Wo); int64_t t = idx / Wo;
    int py = (int)(t % Ho); t /= Ho;
    int co = (int)(t % Cout); int img = (int)(t / Cout);
    const T* ip = in + (int64_t)img * H * W;
    float wl[9];
#pragma unroll
    for (int j = 0; j < 9; ++j) wl[j] = w9[co * 9 + j];
    float s = scale[co], sh = shift[co];
    float mx = -1e30f;
#pragma unroll
    for (int dy = 0; dy < 2; ++dy) {
#pragma unroll
        for (int dx = 0; dx < 2; ++dx) {
            int y0 = 2 * py + dy, x0 = 2 * px + dx;
            float acc = 0.0f;
#pragma unroll
            for (int ky = 0; ky < 3; ++ky) {
                int yy = y0 + ky - 1;
                if (yy < 0 || yy >= H) continue;
#pragma unroll
                for (int kx = 0; kx < 3; ++kx) {
                    int xx = x0 + kx - 1;
                    if (xx < 0 || xx >= W) continue;
                    acc += wl[ky * 3 + kx] * (float)ip[yy * W + xx];
                }
            }
            mx = fmaxf(mx, fmaxf(acc * s + sh, 0.0f));
        }
    }
    out[idx] = (h16)mx;
}

// ---------------------------------------------------------------------------
// Implicit-GEMM conv via WMMA f16->f32, fully specialized per layer shape.
// Block = 256 thr (8 waves), one image, NT*16 output pixels.
// LDS holds NT im2col B-tiles pre-swizzled into B-fragment layout:
//   half index = nt*K*16 + kc*512 + (j>=16)*256 + pixel*16 + (j&15),  j = k%32
// Each wave: one A-fragment (contiguous v16h from pre-tiled weights) feeds NT
// WMMAs (A-operand reuse). Epilogue: BN scale/shift + ReLU, f16 out.
// ---------------------------------------------------------------------------
template <int Cin, int Cout, int H, int W, int NT>
__global__ __launch_bounds__(256)
void conv_wmma_kernel(const h16* __restrict__ actIn, const v16h* __restrict__ Atiles,
                      const float* __restrict__ scale, const float* __restrict__ shift,
                      h16* __restrict__ convOut)
{
    constexpr int K = Cin * 9;
    constexpr int Ktiles = K / 32;
    constexpr int Mtiles = Cout / 16;
    constexpr int HW = H * W;
    constexpr int PIX = 16 * NT;
    constexpr int PSH = (NT == 1) ? 4 : 5;
    __shared__ v16h ldsBv[K * PIX / 16];
    h16* ldsB = (h16*)ldsBv;

    int img = blockIdx.y;
    int base = blockIdx.x * PIX;
    const h16* ip = actIn + (int64_t)img * Cin * HW;

    // ---- stage im2col tile(s) into swizzled LDS (constant div/mod) ----
    for (int e = threadIdx.x; e < K * PIX; e += 256) {
        int p = e & (PIX - 1), k = e >> PSH;
        h16 val = (h16)0.0f;
        int pix = base + p;
        if (pix < HW) {
            int y = pix / W, x = pix - y * W;
            int ci = k / 9, r = k - ci * 9, ky = r / 3, kx = r - ky * 3;
            int yy = y + ky - 1, xx = x + kx - 1;
            if (yy >= 0 && yy < H && xx >= 0 && xx < W)
                val = ip[ci * HW + yy * W + xx];
        }
        int c = k >> 5, j = k & 31, nt = p >> 4;
        ldsB[nt * (K * 16) + (c << 9) + ((j >> 4) << 8) + ((p & 15) << 4) + (j & 15)] = val;
    }
    __syncthreads();

    int wave = threadIdx.x >> 5, lane = threadIdx.x & 31;
    const v16h* Bl = (const v16h*)ldsB;
    int bidx = ((lane >> 4) << 4) + (lane & 15);
    for (int mt = wave; mt < Mtiles; mt += 8) {          // wave-uniform branch
        v8f acc[NT];
#pragma unroll
        for (int nt = 0; nt < NT; ++nt) acc[nt] = (v8f){};
#pragma unroll 3
        for (int kc = 0; kc < Ktiles; ++kc) {
            v16h a = Atiles[(kc * Mtiles + mt) * 32 + lane];
            __builtin_prefetch(&Atiles[((kc + 2) * Mtiles + mt) * 32 + lane], 0, 1);
#pragma unroll
            for (int nt = 0; nt < NT; ++nt) {
                v16h b = Bl[nt * (Ktiles * 32) + kc * 32 + bidx];
                acc[nt] = __builtin_amdgcn_wmma_f32_16x16x32_f16(false, a, false, b,
                                                                 (short)0, acc[nt], false, false);
            }
        }
#pragma unroll
        for (int nt = 0; nt < NT; ++nt) {
            int pix = base + nt * 16 + (lane & 15);
            if (pix < HW) {
                int y = pix / W, x = pix - y * W;
                h16* op = convOut + (int64_t)img * Cout * HW + y * W + x;
                int mbase = mt * 16 + ((lane >> 4) << 3);
#pragma unroll
                for (int r2 = 0; r2 < 8; ++r2) {
                    int m = mbase + r2;
                    float o = fmaxf(acc[nt][r2] * scale[m] + shift[m], 0.0f);
                    op[(int64_t)m * HW] = (h16)o;
                }
            }
        }
    }
}

// ---------------------------------------------------------------------------
// 2x2 max pool (f16 -> f16)
// ---------------------------------------------------------------------------
__global__ void pool_kernel(const h16* __restrict__ in, h16* __restrict__ out,
                            int N, int C, int H, int W)
{
    int Ho = H >> 1, Wo = W >> 1;
    int64_t idx = (int64_t)blockIdx.x * blockDim.x + threadIdx.x;
    int64_t total = (int64_t)N * C * Ho * Wo;
    if (idx >= total) return;
    int px = (int)(idx % Wo); int64_t t = idx / Wo;
    int py = (int)(t % Ho); t /= Ho;
    const h16* ip = in + t * (int64_t)H * W + (2 * py) * W + 2 * px;
    float a = (float)ip[0], b = (float)ip[1], c = (float)ip[W], d = (float)ip[W + 1];
    out[idx] = (h16)fmaxf(fmaxf(a, b), fmaxf(c, d));
}

// ---------------------------------------------------------------------------
// FC / head GEMM via WMMA. B is the contiguous f16 activation [n][K]
// (flatten order matches NCHW storage), so B frag = direct v16h global load.
// Each wave computes TWO adjacent M-tiles sharing one B-fragment (B reuse).
// ---------------------------------------------------------------------------
__global__ __launch_bounds__(256)
void fc_wmma_kernel(const h16* __restrict__ actIn, const v16h* __restrict__ Atiles,
                    const float* __restrict__ scale, const float* __restrict__ shift,
                    h16* __restrict__ out16, float* __restrict__ out32,
                    int K, int Mp, int Mreal, int N, int doRelu)
{
    int Ktiles = K >> 5, Mtiles = Mp >> 4;
    int wave = threadIdx.x >> 5, lane = threadIdx.x & 31;
    int mt0 = (blockIdx.y * 8 + wave) * 2;
    if (mt0 >= Mtiles) return;                            // wave-uniform
    int mt1 = min(mt0 + 1, Mtiles - 1);                   // clamped (idempotent dup)
    int n = blockIdx.x * 16 + (lane & 15);                // N is multiple of 16 here
    const v16h* Bp = (const v16h*)(actIn + (int64_t)n * K + ((lane >> 4) << 4));
    v8f acc0 = {}, acc1 = {};
#pragma unroll 2
    for (int kc = 0; kc < Ktiles; ++kc) {
        v16h b = Bp[kc * 2];
        v16h a0 = Atiles[(kc * Mtiles + mt0) * 32 + lane];
        v16h a1 = Atiles[(kc * Mtiles + mt1) * 32 + lane];
        __builtin_prefetch(&Atiles[((kc + 2) * Mtiles + mt0) * 32 + lane], 0, 1);
        acc0 = __builtin_amdgcn_wmma_f32_16x16x32_f16(false, a0, false, b,
                                                      (short)0, acc0, false, false);
        acc1 = __builtin_amdgcn_wmma_f32_16x16x32_f16(false, a1, false, b,
                                                      (short)0, acc1, false, false);
    }
#pragma unroll
    for (int half = 0; half < 2; ++half) {
        int mt = half ? mt1 : mt0;
        int mbase = mt * 16 + ((lane >> 4) << 3);
#pragma unroll
        for (int r2 = 0; r2 < 8; ++r2) {
            int m = mbase + r2;
            float o = (half ? acc1[r2] : acc0[r2]) * scale[m] + shift[m];
            if (doRelu) o = fmaxf(o, 0.0f);
            if (m < Mreal) {
                if (out16) out16[(int64_t)n * Mp + m] = (h16)o;
                if (out32) out32[(int64_t)n * Mreal + m] = o;
            }
        }
    }
}

// ---------------------------------------------------------------------------
// bbox activations: sigmoid on first 4, tanh on 5th. raw [256][25] -> [1280][5]
// ---------------------------------------------------------------------------
__global__ void bbox_act_kernel(const float* __restrict__ raw, float* __restrict__ bb)
{
    int idx = blockIdx.x * blockDim.x + threadIdx.x;
    if (idx >= 1280 * 5) return;
    int j = idx % 5, q = idx / 5;
    int img = q / 5, ch = q % 5;
    float v = raw[img * 25 + ch * 5 + j];
    bb[idx] = (j < 4) ? (1.0f / (1.0f + expf(-v))) : tanhf(v);
}

// ---------------------------------------------------------------------------
// Bilinear patch extraction, matches reference math. -> f16 patches[1280][40][40]
// ---------------------------------------------------------------------------
__global__ void patch_kernel(const float* __restrict__ x, const float* __restrict__ bb,
                             h16* __restrict__ patches)
{
    int idx = blockIdx.x * blockDim.x + threadIdx.x;
    if (idx >= 1280 * 40 * 40) return;
    int j = idx % 40; int t = idx / 40;
    int i = t % 40; int q = t / 40;
    const float* B5 = bb + q * 5;
    int cx = (int)floorf(B5[0] * 160.0f);
    int cy = (int)floorf(B5[1] * 60.0f);
    int bw = (int)floorf(B5[2] * 160.0f);
    int bh = (int)floorf(B5[3] * 60.0f);
    int x1 = max(0, cx - bw / 2), x2 = min(160, cx + bw / 2);
    int y1 = max(0, cy - bh / 2), y2 = min(60, cy + bh / 2);
    float valid = ((x2 > x1) && (y2 > y1)) ? 1.0f : 0.0f;
    int cw = max(x2 - x1, 1), ch = max(y2 - y1, 1);
    float sy = fmaxf((i + 0.5f) * (float)ch / 40.0f - 0.5f, 0.0f);
    int iy0 = min((int)sy, ch - 1); int iy1 = min(iy0 + 1, ch - 1);
    float fy = sy - (float)iy0;
    float sx = fmaxf((j + 0.5f) * (float)cw / 40.0f - 0.5f, 0.0f);
    int ix0 = min((int)sx, cw - 1); int ix1 = min(ix0 + 1, cw - 1);
    float fx = sx - (float)ix0;
    const float* im = x + (int64_t)(q / 5) * 60 * 160;
    int ay0 = y1 + iy0, ay1 = y1 + iy1, ax0 = x1 + ix0, ax1 = x1 + ix1;
    float v00 = im[ay0 * 160 + ax0], v01 = im[ay0 * 160 + ax1];
    float v10 = im[ay1 * 160 + ax0], v11 = im[ay1 * 160 + ax1];
    float p = (1.0f - fy) * ((1.0f - fx) * v00 + fx * v01)
            + fy * ((1.0f - fx) * v10 + fx * v11);
    patches[idx] = (h16)(p * valid);
}

// ---------------------------------------------------------------------------
// Launch
// ---------------------------------------------------------------------------
extern "C" void kernel_launch(void* const* d_in, const int* in_sizes, int n_in,
                              void* d_out, int out_size, void* d_ws, size_t ws_size,
                              hipStream_t stream)
{
    (void)in_sizes; (void)n_in; (void)out_size; (void)ws_size;
    auto F = [&](int i) { return (const float*)d_in[i]; };
    char* ws = (char*)d_ws;
    h16* P0 = (h16*)(ws + OFF_P0);
    h16* P1 = (h16*)(ws + OFF_P1);
    h16* R  = (h16*)(ws + OFF_R);
    float* SC = (float*)(ws + OFF_SC);
    float* SH = (float*)(ws + OFF_SH);
    float* BBraw = (float*)(ws + OFF_BB);
    float* BB = (float*)(ws + OFF_BB2);
    const int T = 256;
    auto cdiv = [](int64_t a, int64_t b) { return (unsigned)((a + b - 1) / b); };

    // ---- fold BN (+conv/fc bias) into scale/shift ----
    fold_bn_kernel<<<1, T, 0, stream>>>(F(1), F(2), F(3), F(4),   F(18), SC + 0,    SH + 0,    32);
    fold_bn_kernel<<<1, T, 0, stream>>>(F(5), F(6), F(7), F(8),   F(20), SC + 32,   SH + 32,   64);
    fold_bn_kernel<<<1, T, 0, stream>>>(F(9), F(10), F(11), F(12),F(22), SC + 96,   SH + 96,   128);
    fold_bn_kernel<<<1, T, 0, stream>>>(F(13), F(14), F(15), F(16),F(24),SC + 224,  SH + 224,  256);
    fold_bn_kernel<<<2, T, 0, stream>>>(F(26), F(27), F(28), F(29),F(25),SC + 480,  SH + 480,  512);
    head_affine_kernel<<<1, T, 0, stream>>>(F(31), SC + 992,  SH + 992,  25, 32);
    fold_bn_kernel<<<1, T, 0, stream>>>(F(33), F(34), F(35), F(36),F(46),SC + 1024, SH + 1024, 32);
    fold_bn_kernel<<<1, T, 0, stream>>>(F(37), F(38), F(39), F(40),F(48),SC + 1056, SH + 1056, 64);
    fold_bn_kernel<<<1, T, 0, stream>>>(F(41), F(42), F(43), F(44),F(50),SC + 1120, SH + 1120, 128);
    fold_bn_kernel<<<1, T, 0, stream>>>(F(52), F(53), F(54), F(55),F(51),SC + 1248, SH + 1248, 256);
    head_affine_kernel<<<1, T, 0, stream>>>(F(57), SC + 1504, SH + 1504, 36, 48);

    // ---- tile weights into WMMA A-fragment layout (f16) ----
    tile_conv_w_kernel<<<cdiv(18432, T), T, 0, stream>>>(F(19), (h16*)(ws + WT_DC2), 32, 64);
    tile_conv_w_kernel<<<cdiv(73728, T), T, 0, stream>>>(F(21), (h16*)(ws + WT_DC3), 64, 128);
    tile_conv_w_kernel<<<cdiv(294912, T), T, 0, stream>>>(F(23), (h16*)(ws + WT_DC4), 128, 256);
    tile_fc_w_kernel<<<cdiv(3932160, T), T, 0, stream>>>(F(30), (h16*)(ws + WT_DFC), 7680, 512, 512);
    tile_fc_w_kernel<<<cdiv(16384, T), T, 0, stream>>>(F(32), (h16*)(ws + WT_DHD), 512, 25, 32);
    tile_conv_w_kernel<<<cdiv(18432, T), T, 0, stream>>>(F(47), (h16*)(ws + WT_RC2), 32, 64);
    tile_conv_w_kernel<<<cdiv(73728, T), T, 0, stream>>>(F(49), (h16*)(ws + WT_RC3), 64, 128);
    tile_fc_w_kernel<<<cdiv(819200, T), T, 0, stream>>>(F(56), (h16*)(ws + WT_RFC), 3200, 256, 256);
    tile_fc_w_kernel<<<cdiv(12288, T), T, 0, stream>>>(F(58), (h16*)(ws + WT_RHD), 256, 36, 48);

    // ================= detector =================
    conv1_pool_kernel<float><<<cdiv((int64_t)256 * 32 * 30 * 80, T), T, 0, stream>>>(
        F(0), F(17), SC + 0, SH + 0, P0, 256, 32, 60, 160, 30, 80);
    conv_wmma_kernel<32, 64, 30, 80, 2><<<dim3(cdiv(2400, 32), 256), 256, 0, stream>>>(
        P0, (const v16h*)(ws + WT_DC2), SC + 32, SH + 32, R);
    pool_kernel<<<cdiv((int64_t)256 * 64 * 15 * 40, T), T, 0, stream>>>(R, P1, 256, 64, 30, 80);
    conv_wmma_kernel<64, 128, 15, 40, 2><<<dim3(cdiv(600, 32), 256), 256, 0, stream>>>(
        P1, (const v16h*)(ws + WT_DC3), SC + 96, SH + 96, R);
    pool_kernel<<<cdiv((int64_t)256 * 128 * 7 * 20, T), T, 0, stream>>>(R, P0, 256, 128, 15, 40);
    conv_wmma_kernel<128, 256, 7, 20, 1><<<dim3(cdiv(140, 16), 256), 256, 0, stream>>>(
        P0, (const v16h*)(ws + WT_DC4), SC + 224, SH + 224, R);
    pool_kernel<<<cdiv((int64_t)256 * 256 * 3 * 10, T), T, 0, stream>>>(R, P1, 256, 256, 7, 20);
    fc_wmma_kernel<<<dim3(16, 2), 256, 0, stream>>>(
        P1, (const v16h*)(ws + WT_DFC), SC + 480, SH + 480, P0, (float*)nullptr,
        7680, 512, 512, 256, 1);
    fc_wmma_kernel<<<dim3(16, 1), 256, 0, stream>>>(
        P0, (const v16h*)(ws + WT_DHD), SC + 992, SH + 992, (h16*)nullptr, BBraw,
        512, 32, 25, 256, 0);
    bbox_act_kernel<<<cdiv(6400, T), T, 0, stream>>>(BBraw, BB);
    patch_kernel<<<cdiv(2048000, T), T, 0, stream>>>(F(0), BB, P1);

    // ================= recognizer =================
    conv1_pool_kernel<h16><<<cdiv((int64_t)1280 * 32 * 20 * 20, T), T, 0, stream>>>(
        P1, F(45), SC + 1024, SH + 1024, P0, 1280, 32, 40, 40, 20, 20);
    conv_wmma_kernel<32, 64, 20, 20, 2><<<dim3(cdiv(400, 32), 1280), 256, 0, stream>>>(
        P0, (const v16h*)(ws + WT_RC2), SC + 1056, SH + 1056, R);
    pool_kernel<<<cdiv((int64_t)1280 * 64 * 10 * 10, T), T, 0, stream>>>(R, P1, 1280, 64, 20, 20);
    conv_wmma_kernel<64, 128, 10, 10, 2><<<dim3(cdiv(100, 32), 1280), 256, 0, stream>>>(
        P1, (const v16h*)(ws + WT_RC3), SC + 1120, SH + 1120, R);
    pool_kernel<<<cdiv((int64_t)1280 * 128 * 5 * 5, T), T, 0, stream>>>(R, P0, 1280, 128, 10, 10);
    fc_wmma_kernel<<<dim3(80, 1), 256, 0, stream>>>(
        P0, (const v16h*)(ws + WT_RFC), SC + 1248, SH + 1248, P1, (float*)nullptr,
        3200, 256, 256, 1280, 1);
    fc_wmma_kernel<<<dim3(80, 1), 256, 0, stream>>>(
        P1, (const v16h*)(ws + WT_RHD), SC + 1504, SH + 1504, (h16*)nullptr, (float*)d_out,
        256, 48, 36, 1280, 0);
}
